// ModalDecoder_80985903333775
// MI455X (gfx1250) — matrix operand order repaired and checked by reading, not compiled, not run on hardware
//
#include <hip/hip_runtime.h>
#include <hip/hip_bf16.h>

// ---------------------------------------------------------------------------
// ModalDecoder for MI455X (gfx1250, wave32, WMMA).
// Pipeline: cond-GEMM -> LN+FiLM(pack f16 A-frags) -> WMMA GEMM1 (gated GELU)
//           -> WMMA GEMM2 (+bias) -> WMMA GEMM3 (emb-gathered weights) -> out
// GEMM waves are register-blocked (4x2 / 4x4 tiles of 16x16) so each k-step
// issues 16 v_wmma_f32_16x16x32_f16 per 8 fragment loads (~32-43 FLOP/byte
// from L2), keeping the WMMA pipes fed instead of the L2 fabric.
// ---------------------------------------------------------------------------

typedef __attribute__((ext_vector_type(16))) _Float16 v16h;
typedef __attribute__((ext_vector_type(8)))  _Float16 v8h;
typedef __attribute__((ext_vector_type(8)))  float    v8f;

#define C_B    8
#define C_N    4096
#define C_M    32768      // B*N
#define C_D    256
#define C_DN   128
#define C_F    128
#define C_O    8
#define C_H    1024

static __device__ __forceinline__ v8f wmma_f16(v16h a, v16h b, v8f c) {
  return __builtin_amdgcn_wmma_f32_16x16x32_f16(
      /*neg_a=*/false, a, /*neg_b=*/false, b,
      /*c_mod=*/(short)0, c, /*reuse_a=*/false, /*reuse_b=*/false);
}

static __device__ __forceinline__ float gelu_tanh(float x) {
  const float k0 = 0.7978845608028654f;   // sqrt(2/pi)
  const float k1 = 0.044715f;
  float t = tanhf(k0 * (x + k1 * x * x * x));
  return 0.5f * x * (1.0f + t);
}

// Transpose one wave's 16x16 f32 D-tile through LDS so each lane ends up
// holding a contiguous 8-wide run along N (row rr = lane&15, cols c0..c0+7).
// DS ops are in-order per wave; explicit s_wait_dscnt fences the RAW.
static __device__ __forceinline__ void tile_xpose(float* lt, const v8f acc,
                                                  int lane, float out8[8]) {
  const int hi = lane >> 4, mm = lane & 15;
#pragma unroll
  for (int r = 0; r < 8; ++r)
    lt[(r + 8 * hi) * 16 + mm] = acc[r];
  __asm__ volatile("s_wait_dscnt 0" ::: "memory");
  const int rr = lane & 15, c0 = (lane >> 4) * 8;
#pragma unroll
  for (int j = 0; j < 8; ++j)
    out8[j] = lt[rr * 16 + c0 + j];
  __asm__ volatile("s_wait_dscnt 0" ::: "memory");
}

// ---------------------------------------------------------------------------
// K0: ss = ctx @ cond_w + cond_b          [8,128]x[128,512] -> [8,512]
// ---------------------------------------------------------------------------
__global__ void __launch_bounds__(256)
k_cond(const float* __restrict__ ctx, const float* __restrict__ w,
       const float* __restrict__ bias, float* __restrict__ ss) {
  int t = blockIdx.x * 256 + threadIdx.x;        // 0..4095
  int b = t >> 9;                                 // /512
  int c = t & 511;
  float acc = bias[c];
#pragma unroll 4
  for (int k = 0; k < C_DN; ++k)
    acc = fmaf(ctx[b * C_DN + k], w[k * 512 + c], acc);
  ss[t] = acc;
}

// ---------------------------------------------------------------------------
// K1: LayerNorm + FiLM, write h as f16 in packed WMMA A-fragment layout.
// One wave per row (D=256 -> 8 f32 per lane). Packed layout: fragment
// (mtile,ktile) is 32 lanes x 16 f16 contiguous; lane = half*16 + (m&15),
// elem = group*8 + within  (group=k-quarter, half=k-eighth per ISA A-layout).
// ---------------------------------------------------------------------------
__global__ void __launch_bounds__(256)
k_ln_film_pack(const float* __restrict__ x, const float* __restrict__ ss,
               _Float16* __restrict__ hp) {
  const int wave = threadIdx.x >> 5, lane = threadIdx.x & 31;
  const int row = blockIdx.x * 8 + wave;          // 0..32767
  const int b = row >> 12;                        // row / 4096
  const float* xr = x + (size_t)row * C_D + lane * 8;
  float v[8];
  float s = 0.f, s2 = 0.f;
#pragma unroll
  for (int j = 0; j < 8; ++j) { v[j] = xr[j]; s += v[j]; s2 += v[j] * v[j]; }
#pragma unroll
  for (int m = 16; m > 0; m >>= 1) {
    s  += __shfl_xor(s,  m, 32);
    s2 += __shfl_xor(s2, m, 32);
  }
  const float mu   = s * (1.0f / C_D);
  const float var  = s2 * (1.0f / C_D) - mu * mu;
  const float rstd = rsqrtf(var + 1e-5f);
  const float* scl = ss + b * 512;
  const float* sft = scl + C_D;
  v8h o;
#pragma unroll
  for (int j = 0; j < 8; ++j) {
    const int d = lane * 8 + j;
    float hv = (v[j] - mu) * rstd * (1.0f + scl[d]) + sft[d];
    o[j] = (_Float16)hv;
  }
  const int mtile = row >> 4, mm = row & 15;
  const int kt   = lane >> 2;            // (lane*8)>>5
  const int grp  = (lane >> 1) & 1;
  const int half = lane & 1;
  size_t off = ((size_t)(mtile * 8 + kt) * 32 + (half * 16 + mm)) * 16 + grp * 8;
  *(v8h*)(hp + off) = o;
}

// ---------------------------------------------------------------------------
// K2: pack fp32 weight [K x Ncols] (row-major) into f16 B-fragment layout.
// Fragment (ntile,ktile): lane = khalf*16 + (n&15), elem i -> K = khalf*8+(i&7)+(i>>3)*16.
// ---------------------------------------------------------------------------
__global__ void __launch_bounds__(256)
k_pack_b(const float* __restrict__ src, _Float16* __restrict__ dst,
         int K, int Ncols) {
  const int t = blockIdx.x * 256 + threadIdx.x;
  const int lane = t & 31, frag = t >> 5;
  const int KT = K >> 5;
  const int kt = frag % KT, ntile = frag / KT;
  const int khalf = lane >> 4, nn = lane & 15;
  const int n = ntile * 16 + nn;
  v16h val;
#pragma unroll
  for (int i = 0; i < 16; ++i) {
    const int k = kt * 32 + khalf * 8 + (i & 7) + (i >> 3) * 16;
    val[i] = (_Float16)src[(size_t)k * Ncols + n];
  }
  *(v16h*)(dst + (size_t)t * 16) = val;
}

// K2b: pack gathered emb[idx] -> W2[d, f*8+o] into B-fragment layout f16.
__global__ void __launch_bounds__(256)
k_pack_w2(const float* __restrict__ emb, const int* __restrict__ idx,
          _Float16* __restrict__ dst) {
  const int t = blockIdx.x * 256 + threadIdx.x;   // K=256 -> KT=8, Ncols=1024
  const int lane = t & 31, frag = t >> 5;
  const int kt = frag & 7, ntile = frag >> 3;
  const int khalf = lane >> 4, nn = lane & 15;
  const int n = ntile * 16 + nn;
  const int f = n >> 3, o = n & 7;
  const long long row = (long long)idx[f];
  v16h val;
#pragma unroll
  for (int i = 0; i < 16; ++i) {
    const int d = kt * 32 + khalf * 8 + (i & 7) + (i >> 3) * 16;
    val[i] = (_Float16)emb[(size_t)row * (C_D * C_O) + (size_t)d * C_O + o];
  }
  *(v16h*)(dst + (size_t)t * 16) = val;
}

// ---------------------------------------------------------------------------
// K3: GEMM1  uv = h @ W_in ; g = gelu(u)*v, stored as packed A-frags (K=1024).
// Wave computes a 64x32 g-panel: u[4][2] + v[4][2] accumulators; per k-step
// 4 A + 4 B fragment loads feed 16 WMMAs (A shared between u and v).
// ---------------------------------------------------------------------------
__global__ void __launch_bounds__(256)
k_gemm1(const _Float16* __restrict__ A, const _Float16* __restrict__ Bw,
        const float* __restrict__ bias, _Float16* __restrict__ Gp) {
  __shared__ float lds[8 * 256];
  const int wave = threadIdx.x >> 5, lane = threadIdx.x & 31;
  const int w = blockIdx.x * 8 + wave;
  const int nPanel = w & 31;                 // 32 panels of 32 g-columns
  const int mPanel = w >> 5;                 // 512 panels of 64 rows
  const _Float16* ap[4];
#pragma unroll
  for (int i = 0; i < 4; ++i)
    ap[i] = A + (size_t)(mPanel * 4 + i) * 8 * 512 + lane * 16;
  const _Float16* bu[2]; const _Float16* bv[2];
#pragma unroll
  for (int j = 0; j < 2; ++j) {
    const int ntg = nPanel * 2 + j;
    bu[j] = Bw + (size_t)ntg * 8 * 512 + lane * 16;
    bv[j] = Bw + (size_t)(ntg + 64) * 8 * 512 + lane * 16;
  }
  v8f cu[4][2] = {}; v8f cv[4][2] = {};
#pragma unroll
  for (int kt = 0; kt < 8; ++kt) {
    __builtin_prefetch(ap[0] + (size_t)(kt + 1) * 512, 0, 1);
    v16h a[4], pu[2], pv[2];
#pragma unroll
    for (int i = 0; i < 4; ++i) a[i] = *(const v16h*)(ap[i] + (size_t)kt * 512);
#pragma unroll
    for (int j = 0; j < 2; ++j) {
      pu[j] = *(const v16h*)(bu[j] + (size_t)kt * 512);
      pv[j] = *(const v16h*)(bv[j] + (size_t)kt * 512);
    }
#pragma unroll
    for (int i = 0; i < 4; ++i)
#pragma unroll
      for (int j = 0; j < 2; ++j) {
        cu[i][j] = wmma_f16(a[i], pu[j], cu[i][j]);
        cv[i][j] = wmma_f16(a[i], pv[j], cv[i][j]);
      }
  }
  const int nn = lane & 15;
  float* lt = &lds[wave * 256];
#pragma unroll
  for (int i = 0; i < 4; ++i) {
#pragma unroll
    for (int j = 0; j < 2; ++j) {
      const int ntg = nPanel * 2 + j;
      const float ub = bias[ntg * 16 + nn];
      const float vb = bias[C_H + ntg * 16 + nn];
      v8f g;
#pragma unroll
      for (int r = 0; r < 8; ++r)
        g[r] = gelu_tanh(cu[i][j][r] + ub) * (cv[i][j][r] + vb);
      float out8[8];
      tile_xpose(lt, g, lane, out8);
      v8h o;
#pragma unroll
      for (int q = 0; q < 8; ++q) o[q] = (_Float16)out8[q];
      const int mt = mPanel * 4 + i;
      size_t off = ((size_t)(mt * 32 + (ntg >> 1)) * 32 + lane) * 16 + (ntg & 1) * 8;
      *(v8h*)(Gp + off) = o;
    }
  }
}

// ---------------------------------------------------------------------------
// K4: GEMM2  y = g @ W_out + b_out, stored as packed A-frags (K=256).
// Wave computes 64x64: acc[4][4]; 8 fragment loads -> 16 WMMAs per k-step.
// ---------------------------------------------------------------------------
__global__ void __launch_bounds__(256)
k_gemm2(const _Float16* __restrict__ A, const _Float16* __restrict__ Bw,
        const float* __restrict__ bias, _Float16* __restrict__ Yp) {
  __shared__ float lds[8 * 256];
  const int wave = threadIdx.x >> 5, lane = threadIdx.x & 31;
  const int w = blockIdx.x * 8 + wave;
  const int nPanel = w & 3;                  // 4 panels of 64 columns
  const int mPanel = w >> 2;                 // 512 panels of 64 rows
  const _Float16* ap[4]; const _Float16* bp[4];
#pragma unroll
  for (int i = 0; i < 4; ++i) {
    ap[i] = A  + (size_t)(mPanel * 4 + i) * 32 * 512 + lane * 16;
    bp[i] = Bw + (size_t)(nPanel * 4 + i) * 32 * 512 + lane * 16;
  }
  v8f acc[4][4] = {};
#pragma unroll 2
  for (int kt = 0; kt < 32; ++kt) {
    __builtin_prefetch(ap[0] + (size_t)(kt + 1) * 512, 0, 1);
    v16h a[4], b[4];
#pragma unroll
    for (int i = 0; i < 4; ++i) a[i] = *(const v16h*)(ap[i] + (size_t)kt * 512);
#pragma unroll
    for (int j = 0; j < 4; ++j) b[j] = *(const v16h*)(bp[j] + (size_t)kt * 512);
#pragma unroll
    for (int i = 0; i < 4; ++i)
#pragma unroll
      for (int j = 0; j < 4; ++j)
        acc[i][j] = wmma_f16(a[i], b[j], acc[i][j]);
  }
  const int nn = lane & 15;
  float* lt = &lds[wave * 256];
#pragma unroll
  for (int i = 0; i < 4; ++i) {
#pragma unroll
    for (int j = 0; j < 4; ++j) {
      const int ntile = nPanel * 4 + j;
      const float yb = bias[ntile * 16 + nn];
      v8f c = acc[i][j];
#pragma unroll
      for (int r = 0; r < 8; ++r) c[r] += yb;
      float out8[8];
      tile_xpose(lt, c, lane, out8);
      v8h o;
#pragma unroll
      for (int q = 0; q < 8; ++q) o[q] = (_Float16)out8[q];
      const int mt = mPanel * 4 + i;
      size_t off = ((size_t)(mt * 8 + (ntile >> 1)) * 32 + lane) * 16 + (ntile & 1) * 8;
      *(v8h*)(Yp + off) = o;
    }
  }
}

// ---------------------------------------------------------------------------
// K5: GEMM3  out = y @ W2  -> fp32 [32768, 1024] (row m, col f*8+o).
// Wave computes 64x64: acc[4][4]; 8 fragment loads -> 16 WMMAs per k-step.
// ---------------------------------------------------------------------------
__global__ void __launch_bounds__(256)
k_gemm3(const _Float16* __restrict__ A, const _Float16* __restrict__ Bw,
        float* __restrict__ out) {
  __shared__ float lds[8 * 256];
  const int wave = threadIdx.x >> 5, lane = threadIdx.x & 31;
  const int w = blockIdx.x * 8 + wave;
  const int nPanel = w & 15;                 // 16 panels of 64 columns
  const int mPanel = w >> 4;                 // 512 panels of 64 rows
  const _Float16* ap[4]; const _Float16* bp[4];
#pragma unroll
  for (int i = 0; i < 4; ++i) {
    ap[i] = A  + (size_t)(mPanel * 4 + i) * 8 * 512 + lane * 16;
    bp[i] = Bw + (size_t)(nPanel * 4 + i) * 8 * 512 + lane * 16;
  }
  v8f acc[4][4] = {};
#pragma unroll
  for (int kt = 0; kt < 8; ++kt) {
    __builtin_prefetch(ap[0] + (size_t)(kt + 1) * 512, 0, 1);
    v16h a[4], b[4];
#pragma unroll
    for (int i = 0; i < 4; ++i) a[i] = *(const v16h*)(ap[i] + (size_t)kt * 512);
#pragma unroll
    for (int j = 0; j < 4; ++j) b[j] = *(const v16h*)(bp[j] + (size_t)kt * 512);
#pragma unroll
    for (int i = 0; i < 4; ++i)
#pragma unroll
      for (int j = 0; j < 4; ++j)
        acc[i][j] = wmma_f16(a[i], b[j], acc[i][j]);
  }
  float* lt = &lds[wave * 256];
  const int rr = lane & 15, c0 = (lane >> 4) * 8;
#pragma unroll
  for (int i = 0; i < 4; ++i) {
#pragma unroll
    for (int j = 0; j < 4; ++j) {
      float out8[8];
      tile_xpose(lt, acc[i][j], lane, out8);
      const int mt = mPanel * 4 + i, ntile = nPanel * 4 + j;
      float* op = out + (size_t)(mt * 16 + rr) * 1024 + ntile * 16 + c0;
      float4 lo = make_float4(out8[0], out8[1], out8[2], out8[3]);
      float4 hi = make_float4(out8[4], out8[5], out8[6], out8[7]);
      *(float4*)op       = lo;
      *(float4*)(op + 4) = hi;
    }
  }
}

// ---------------------------------------------------------------------------
extern "C" void kernel_launch(void* const* d_in, const int* in_sizes, int n_in,
                              void* d_out, int out_size, void* d_ws, size_t ws_size,
                              hipStream_t stream) {
  (void)in_sizes; (void)n_in; (void)out_size; (void)ws_size;
  const float* x         = (const float*)d_in[0];
  const float* ctx       = (const float*)d_in[1];
  const int*   idx       = (const int*)d_in[2];
  const float* cond_w    = (const float*)d_in[3];
  const float* cond_b    = (const float*)d_in[4];
  const float* ffn_w_in  = (const float*)d_in[5];
  const float* ffn_b_in  = (const float*)d_in[6];
  const float* ffn_w_out = (const float*)d_in[7];
  const float* ffn_b_out = (const float*)d_in[8];
  const float* emb       = (const float*)d_in[9];
  float* out = (float*)d_out;
  char*  ws  = (char*)d_ws;

  // Workspace layout (bytes, generously aligned)
  size_t off_ss   = 0;                                     // 8*512 f32
  size_t off_h    = 64 * 1024;                             // 32768*256 f16
  size_t off_win  = off_h   + (size_t)C_M * C_D * 2;       // 256*2048 f16
  size_t off_wout = off_win + (size_t)C_D * 2 * C_H * 2;   // 1024*256 f16
  size_t off_w2   = off_wout+ (size_t)C_H * C_D * 2;       // 256*1024 f16
  size_t off_g    = off_w2  + (size_t)C_D * (C_F * C_O) * 2; // 32768*1024 f16
  size_t off_y    = off_g   + (size_t)C_M * C_H * 2;       // 32768*256 f16

  float*    ss  = (float*)(ws + off_ss);
  _Float16* hp  = (_Float16*)(ws + off_h);
  _Float16* wip = (_Float16*)(ws + off_win);
  _Float16* wop = (_Float16*)(ws + off_wout);
  _Float16* w2p = (_Float16*)(ws + off_w2);
  _Float16* gp  = (_Float16*)(ws + off_g);
  _Float16* yp  = (_Float16*)(ws + off_y);

  // K0: conditioning GEMM (tiny)
  k_cond<<<16, 256, 0, stream>>>(ctx, cond_w, cond_b, ss);
  // Weight packers (same stream serializes everything)
  k_pack_b<<<128, 256, 0, stream>>>(ffn_w_in,  wip, C_D, 2 * C_H);   // 8*128 frags
  k_pack_b<<<64,  256, 0, stream>>>(ffn_w_out, wop, C_H, C_D);       // 32*16 frags
  k_pack_w2<<<64, 256, 0, stream>>>(emb, idx, w2p);                  // 8*64 frags
  // K1: LayerNorm + FiLM -> packed f16 activations
  k_ln_film_pack<<<C_M / 8, 256, 0, stream>>>(x, ss, hp);
  // K3: GEMM1 + gated GELU  (512 mPanels x 32 nPanels waves, 8 waves/block)
  k_gemm1<<<(512 * 32) / 8, 256, 0, stream>>>(hp, wip, ffn_b_in, gp);
  // K4: GEMM2 + bias        (512 x 4 waves)
  k_gemm2<<<(512 * 4) / 8, 256, 0, stream>>>(gp, wop, ffn_b_out, yp);
  // K5: GEMM3 -> fp32 out   (512 x 16 waves)
  k_gemm3<<<(512 * 16) / 8, 256, 0, stream>>>(yp, w2p, out);
}
